// GroupedQueryAttention_78872779424176
// MI455X (gfx1250) — matrix operand order repaired and checked by reading, not compiled, not run on hardware
//
#include <hip/hip_runtime.h>

// ---------------------------------------------------------------------------
// Problem constants (B, S, D, G) = (4, 2048, 1024, 8), DK = 128
// ---------------------------------------------------------------------------
constexpr int Bn  = 4;
constexpr int Sn  = 2048;
constexpr int Dn  = 1024;
constexpr int Gn  = 8;
constexpr int DKn = 128;
constexpr int Mrows = Bn * Sn;            // 8192 tokens
constexpr float QSCALE = 0.08838834764831845f;   // 1/sqrt(128), folded into q
constexpr float LOG2E  = 1.4426950408889634f;

typedef __attribute__((ext_vector_type(16))) __bf16 v16bf;
typedef __attribute__((ext_vector_type(8)))  __bf16 v8bf;
typedef __attribute__((ext_vector_type(8)))  float  v8f;
typedef __attribute__((ext_vector_type(4)))  int    v4i;

// Pointer-to-v4i in explicit address spaces (global=1, LDS=3); the builtin's
// prototype (from the round-2 diagnostic) wants typed v4i pointers.
typedef __attribute__((address_space(1))) v4i* as1_v4i;
typedef __attribute__((address_space(3))) v4i* as3_v4i;

// ---------------------------------------------------------------------------
// WMMA helpers (V_WMMA_F32_16X16X32_BF16, wave32)
// ---------------------------------------------------------------------------
__device__ __forceinline__ v8f wmma_bf16(v16bf a, v16bf b, v8f c) {
  // 8 args: (neg_a, A, neg_b, B, c_mod, C, reuse_a, reuse_b)
  return __builtin_amdgcn_wmma_f32_16x16x32_bf16(false, a, false, b,
                                                 (short)0, c, false, false);
}

// A-fragment (16x32, row-major source): lane l -> row m=l&15, h=l>>4.
// elements 0..7  <- mem[base + 0 .. 7]   (K = h*8 + e)
// elements 8..15 <- mem[base + 16 .. 23] (K = 16 + h*8 + e)
// caller passes base = row_ptr + kb + h*8
__device__ __forceinline__ v16bf load_a_frag(const __bf16* base) {
  v8bf lo = *reinterpret_cast<const v8bf*>(base);
  v8bf hi = *reinterpret_cast<const v8bf*>(base + 16);
  return __builtin_shufflevector(lo, hi, 0,1,2,3,4,5,6,7,8,9,10,11,12,13,14,15);
}

// B-fragment (32x16, source transposed to N x K row-major): lane l -> col
// n=l&15, h=l>>4; elements 0..7 <- mem[n*ldk + kb + h*8 + e] (K = h*8 + e),
// elements 8..15 <- +16 (K = 16 + h*8 + e)
__device__ __forceinline__ v16bf load_b_frag(const __bf16* base) {
  v8bf lo = *reinterpret_cast<const v8bf*>(base);
  v8bf hi = *reinterpret_cast<const v8bf*>(base + 16);
  return __builtin_shufflevector(lo, hi, 0,1,2,3,4,5,6,7,8,9,10,11,12,13,14,15);
}

__device__ __forceinline__ float fast_exp2(float x) {
#if __has_builtin(__builtin_amdgcn_exp2f)
  return __builtin_amdgcn_exp2f(x);
#else
  return exp2f(x);
#endif
}

__device__ __forceinline__ v8f v8f_zero() {
  v8f z = {0.f, 0.f, 0.f, 0.f, 0.f, 0.f, 0.f, 0.f};
  return z;
}

// ---------------------------------------------------------------------------
// Async global -> LDS copy (GLOBAL_LOAD_ASYNC_TO_LDS_B128, ASYNCcnt tracked).
// Pointers are produced via integer casts: a generic pointer to LDS carries
// the LDS byte offset in its low 32 bits (flat aperture rule), and a generic
// global pointer equals its AS(1) address.
// ---------------------------------------------------------------------------
#if defined(__has_builtin)
#if __has_builtin(__builtin_amdgcn_global_load_async_to_lds_b128)
#define ASYNC_VIA_BUILTIN 1
#endif
#endif
#ifndef ASYNC_VIA_BUILTIN
#define ASYNC_VIA_BUILTIN 0
#endif

__device__ __forceinline__ void async_cp16(__bf16* ldst, const __bf16* gsrc) {
#if ASYNC_VIA_BUILTIN
  as1_v4i g = (as1_v4i)(unsigned long long)(uintptr_t)gsrc;
  as3_v4i l = (as3_v4i)(unsigned)(uintptr_t)ldst;
  __builtin_amdgcn_global_load_async_to_lds_b128(g, l, 0, 0);
#else
  unsigned lds = (unsigned)(uintptr_t)ldst;
  unsigned long long ga = (unsigned long long)(uintptr_t)gsrc;
  asm volatile("global_load_async_to_lds_b128 %0, %1, off"
               :: "v"(lds), "v"(ga) : "memory");
#endif
}

__device__ __forceinline__ void wait_async0() {
#if defined(__has_builtin) && __has_builtin(__builtin_amdgcn_s_wait_asynccnt)
  __builtin_amdgcn_s_wait_asynccnt(0);
#else
  asm volatile("s_wait_asynccnt 0" ::: "memory");
#endif
}

// ---------------------------------------------------------------------------
// Elementwise f32 -> bf16 convert
// ---------------------------------------------------------------------------
__global__ __launch_bounds__(256) void cvt_bf16(const float* __restrict__ in,
                                                __bf16* __restrict__ out, int n) {
  int i = blockIdx.x * 256 + threadIdx.x;
  if (i < n) out[i] = (__bf16)in[i];
}

// Transpose + convert: in slice is [K][N] f32 row-major -> out slice [N][K] bf16.
__global__ __launch_bounds__(256) void transpose_cvt(const float* __restrict__ in,
                                                     __bf16* __restrict__ out,
                                                     int K, int N) {
  size_t slice = (size_t)K * N;
  const float* ip = in + slice * blockIdx.z;
  __bf16* op = out + slice * blockIdx.z;
  int i = blockIdx.x * 256 + threadIdx.x;
  if (i < K * N) {
    int k = i / N, n = i - k * N;
    op[(size_t)n * K + k] = (__bf16)ip[i];
  }
}

// ---------------------------------------------------------------------------
// Tiled WMMA GEMM:  C(M x N) = A(M x K) * Bt(N x K)^T + bias
//   block = 256 threads = 8 waves arranged 4(M) x 2(N); block tile 128x128
//   wave tile 32 x 64 = 2 x 4 grid of 16x16 WMMA accumulators
// MODE 0: q-projection   -> qh (B,G,S,DK) bf16, scaled by 1/sqrt(DK)
// MODE 1: kv-projection  -> kh (B,G,S,DK) bf16 ; vT (B,G,DK,S) bf16 (g=blockIdx.z)
// MODE 2: o-projection   -> outF (M x N) f32
// ---------------------------------------------------------------------------
template <int MODE>
__global__ __launch_bounds__(256) void gemm_wmma(const __bf16* __restrict__ A,
                                                 const __bf16* __restrict__ Bt,
                                                 const float* __restrict__ bias,
                                                 float* __restrict__ outF,
                                                 __bf16* __restrict__ out0,
                                                 __bf16* __restrict__ out1) {
  constexpr int K = Dn;  // 1024 for all three GEMMs
  const int wave = threadIdx.x >> 5;
  const int lane = threadIdx.x & 31;
  const int lr = lane & 15;
  const int h  = lane >> 4;
  const int g  = (MODE == 1) ? blockIdx.z : 0;

  const int m_base = blockIdx.x * 128 + (wave & 3) * 32;
  const int n_base = blockIdx.y * 128 + (wave >> 2) * 64;

  const __bf16* Bt_  = Bt + (size_t)g * 2 * DKn * K;   // per-group weight slice
  const float*  bias_ = bias + g * 2 * DKn;

  v8f acc[2][4];
#pragma unroll
  for (int i = 0; i < 2; i++)
#pragma unroll
    for (int j = 0; j < 4; j++) acc[i][j] = v8f_zero();

  for (int kb = 0; kb < K; kb += 32) {
    v16bf af[2], bf[4];
#pragma unroll
    for (int i = 0; i < 2; i++) {
      const __bf16* p = A + (size_t)(m_base + i * 16 + lr) * K + kb + h * 8;
      af[i] = load_a_frag(p);
    }
#pragma unroll
    for (int j = 0; j < 4; j++) {
      const __bf16* p = Bt_ + (size_t)(n_base + j * 16 + lr) * K + kb + h * 8;
      bf[j] = load_b_frag(p);
    }
#pragma unroll
    for (int i = 0; i < 2; i++)
#pragma unroll
      for (int j = 0; j < 4; j++) acc[i][j] = wmma_bf16(af[i], bf[j], acc[i][j]);
  }

  // C layout: element r of lane l -> row M = h*8 + r, col N = lr
#pragma unroll
  for (int i = 0; i < 2; i++)
#pragma unroll
    for (int j = 0; j < 4; j++)
#pragma unroll
      for (int r = 0; r < 8; r++) {
        int row = m_base + i * 16 + h * 8 + r;
        int col = n_base + j * 16 + lr;
        float v = acc[i][j][r] + bias_[col];
        if constexpr (MODE == 0) {
          v *= QSCALE;
          int b_ = row >> 11, s_ = row & (Sn - 1);
          int g_ = col >> 7, dk = col & (DKn - 1);
          out0[(((size_t)(b_ * Gn + g_) * Sn) + s_) * DKn + dk] = (__bf16)v;
        } else if constexpr (MODE == 1) {
          int b_ = row >> 11, s_ = row & (Sn - 1);
          if (col < DKn)
            out0[(((size_t)(b_ * Gn + g) * Sn) + s_) * DKn + col] = (__bf16)v;
          else
            out1[(((size_t)(b_ * Gn + g) * DKn) + (col - DKn)) * Sn + s_] = (__bf16)v;
        } else {
          outF[(size_t)row * Dn + col] = v;
        }
      }
}

// ---------------------------------------------------------------------------
// Flash attention, one (b,g) per blockIdx.y, 128 query rows per block,
// each wave owns 16 query rows. Keys streamed in 64-wide chunks that are
// staged cooperatively into LDS with async global->LDS copies (double
// buffered, ASYNCcnt + block barrier), removing the 8x per-wave redundancy.
//   scores: 4x4 = 16 WMMAs / chunk; ctx: 2x8 = 16 WMMAs / chunk
// P is bounced through LDS to convert C-layout -> A-layout.
// ---------------------------------------------------------------------------
__global__ __launch_bounds__(256) void attn_flash(const __bf16* __restrict__ qh,
                                                  const __bf16* __restrict__ kh,
                                                  const __bf16* __restrict__ vT,
                                                  __bf16* __restrict__ comb) {
  constexpr int KTS = 136;  // K-tile row stride (128 + 8 pad) -> bank spread
  constexpr int VTS = 72;   // V-tile row stride (64 + 8 pad)  -> bank spread
  __shared__ __attribute__((aligned(16))) __bf16 Kt[2][64 * KTS];
  __shared__ __attribute__((aligned(16))) __bf16 Vt[2][128 * VTS];
  __shared__ __attribute__((aligned(16))) __bf16 pbuf[8][16][72];

  const int tid  = threadIdx.x;
  const int wave = tid >> 5;
  const int lane = tid & 31;
  const int lr = lane & 15;
  const int h  = lane >> 4;

  const int bg = blockIdx.y;
  const int b  = bg >> 3;
  const int g  = bg & 7;

  const __bf16* qb = qh + (size_t)bg * Sn * DKn;
  const __bf16* kb = kh + (size_t)bg * Sn * DKn;
  const __bf16* vb = vT + (size_t)bg * DKn * Sn;

  const int q0 = blockIdx.x * 128 + wave * 16;

  // Stage one 64-key chunk: K rows 64 x 256B (16 units), V rows 128 x 128B (8
  // units); 2048 16B units over 256 threads = 8 async copies per thread.
  auto stage = [&](int kc, int buf) {
#pragma unroll
    for (int it = 0; it < 4; it++) {
      int u = tid + it * 256;           // 0..1023
      int r = u >> 4, c = u & 15;       // K: 64 rows x 16 units
      async_cp16(&Kt[buf][r * KTS + c * 8],
                 kb + (size_t)(kc + r) * DKn + c * 8);
    }
#pragma unroll
    for (int it = 0; it < 4; it++) {
      int u = tid + it * 256;           // 0..1023
      int r = u >> 3, c = u & 7;        // V: 128 rows x 8 units
      async_cp16(&Vt[buf][r * VTS + c * 8],
                 vb + (size_t)r * Sn + kc + c * 8);
    }
  };

  // Resident q A-fragments covering full head dim (K = 128 = 4 x 32)
  v16bf qf[4];
#pragma unroll
  for (int kk = 0; kk < 4; kk++) {
    const __bf16* p = qb + (size_t)(q0 + lr) * DKn + kk * 32 + h * 8;
    qf[kk] = load_a_frag(p);
  }

  v8f acc[8];
#pragma unroll
  for (int j = 0; j < 8; j++) acc[j] = v8f_zero();
  float mrow[8], lrow[8];
#pragma unroll
  for (int r = 0; r < 8; r++) { mrow[r] = -1e30f; lrow[r] = 0.f; }

  stage(0, 0);
  int cur = 0;

  for (int kc = 0; kc < Sn; kc += 64) {
    wait_async0();        // this wave's copies into Kt/Vt[cur] have landed
    __syncthreads();      // everyone's copies landed; prev compute all done
    if (kc + 64 < Sn) stage(kc + 64, cur ^ 1);

    const __bf16* Ktc = Kt[cur];
    const __bf16* Vtc = Vt[cur];

    // ---- scores: 4 tiles of 16 keys, K-dim = 128 ----
    v8f sc[4];
#pragma unroll
    for (int t = 0; t < 4; t++) sc[t] = v8f_zero();
#pragma unroll
    for (int t = 0; t < 4; t++)
#pragma unroll
      for (int kk = 0; kk < 4; kk++) {
        const __bf16* p = Ktc + (t * 16 + lr) * KTS + kk * 32 + h * 8;
        sc[t] = wmma_bf16(qf[kk], load_b_frag(p), sc[t]);
      }

    // ---- online softmax; row M = h*8+r lives in the 16 lanes of one half ----
#pragma unroll
    for (int r = 0; r < 8; r++) {
      float cm = fmaxf(fmaxf(sc[0][r], sc[1][r]), fmaxf(sc[2][r], sc[3][r]));
      cm = fmaxf(cm, __shfl_xor(cm, 1, 32));
      cm = fmaxf(cm, __shfl_xor(cm, 2, 32));
      cm = fmaxf(cm, __shfl_xor(cm, 4, 32));
      cm = fmaxf(cm, __shfl_xor(cm, 8, 32));
      float mn = fmaxf(mrow[r], cm);
      float alpha = fast_exp2((mrow[r] - mn) * LOG2E);
      mrow[r] = mn;
      float rs = 0.f;
#pragma unroll
      for (int t = 0; t < 4; t++) {
        float pv = fast_exp2((sc[t][r] - mn) * LOG2E);
        sc[t][r] = pv;
        rs += pv;
      }
      rs += __shfl_xor(rs, 1, 32);
      rs += __shfl_xor(rs, 2, 32);
      rs += __shfl_xor(rs, 4, 32);
      rs += __shfl_xor(rs, 8, 32);
      lrow[r] = lrow[r] * alpha + rs;
#pragma unroll
      for (int j = 0; j < 8; j++) acc[j][r] *= alpha;
      // stash P (C-layout) into LDS
#pragma unroll
      for (int t = 0; t < 4; t++)
        pbuf[wave][h * 8 + r][t * 16 + lr] = (__bf16)sc[t][r];
    }

    // ---- reload P in A-layout, accumulate ctx += P(16x64) @ V(64x128) ----
    v16bf pf[2];
#pragma unroll
    for (int kk = 0; kk < 2; kk++)
      pf[kk] = load_a_frag(&pbuf[wave][lr][kk * 32 + h * 8]);
#pragma unroll
    for (int j = 0; j < 8; j++)
#pragma unroll
      for (int kk = 0; kk < 2; kk++) {
        const __bf16* p = Vtc + (j * 16 + lr) * VTS + kk * 32 + h * 8;
        acc[j] = wmma_bf16(pf[kk], load_b_frag(p), acc[j]);
      }
    cur ^= 1;
  }

  // ---- normalize and write ctx into combined (B,S,D) bf16 ----
#pragma unroll
  for (int r = 0; r < 8; r++) {
    float inv = 1.0f / lrow[r];
    int srow = q0 + h * 8 + r;
    size_t orow = ((size_t)b * Sn + srow) * Dn + g * DKn;
#pragma unroll
    for (int j = 0; j < 8; j++)
      comb[orow + j * 16 + lr] = (__bf16)(acc[j][r] * inv);
  }
}

// ---------------------------------------------------------------------------
// Host-side launcher
// ---------------------------------------------------------------------------
extern "C" void kernel_launch(void* const* d_in, const int* in_sizes, int n_in,
                              void* d_out, int out_size, void* d_ws, size_t ws_size,
                              hipStream_t stream) {
  const float* Q     = (const float*)d_in[0];
  const float* K     = (const float*)d_in[1];
  // d_in[2] = V input: unused by the reference computation
  const float* Wq_w  = (const float*)d_in[3];
  const float* Wq_b  = (const float*)d_in[4];
  const float* Wkv_w = (const float*)d_in[5];
  const float* Wkv_b = (const float*)d_in[6];
  const float* Wo_w  = (const float*)d_in[7];
  const float* Wo_b  = (const float*)d_in[8];
  float* out = (float*)d_out;

  char* ws = (char*)d_ws;
  // Workspace layout (bytes). comb aliases Qbf (Qbf dead after q-proj).
  constexpr size_t SZ_ACT = (size_t)Mrows * Dn * 2;        // 16 MiB
  constexpr size_t OFF_QBF  = 0;
  constexpr size_t OFF_KBF  = OFF_QBF  + SZ_ACT;           // 16 MiB
  constexpr size_t OFF_WQT  = OFF_KBF  + SZ_ACT;           // 32 MiB
  constexpr size_t OFF_WKVT = OFF_WQT  + (size_t)Dn * Dn * 2;            // 34 MiB
  constexpr size_t OFF_WOT  = OFF_WKVT + (size_t)Gn * 2 * DKn * Dn * 2;  // 38 MiB
  constexpr size_t OFF_QH   = OFF_WOT  + (size_t)Dn * Dn * 2;            // 40 MiB
  constexpr size_t OFF_KH   = OFF_QH   + SZ_ACT;           // 56 MiB
  constexpr size_t OFF_VT   = OFF_KH   + SZ_ACT;           // 72 MiB (end 88 MiB)

  __bf16* Qbf  = (__bf16*)(ws + OFF_QBF);
  __bf16* Kbf  = (__bf16*)(ws + OFF_KBF);
  __bf16* Wqt  = (__bf16*)(ws + OFF_WQT);
  __bf16* Wkvt = (__bf16*)(ws + OFF_WKVT);
  __bf16* Wot  = (__bf16*)(ws + OFF_WOT);
  __bf16* qhp  = (__bf16*)(ws + OFF_QH);
  __bf16* khp  = (__bf16*)(ws + OFF_KH);
  __bf16* vTp  = (__bf16*)(ws + OFF_VT);
  __bf16* comb = (__bf16*)(ws + OFF_QBF);   // alias: Qbf dead after q-proj

  const int nAct = Mrows * Dn;  // 8,388,608

  // 1) input converts + weight transpose/convert to bf16 (N,K) layout
  cvt_bf16<<<dim3(nAct / 256), 256, 0, stream>>>(Q, Qbf, nAct);
  cvt_bf16<<<dim3(nAct / 256), 256, 0, stream>>>(K, Kbf, nAct);
  transpose_cvt<<<dim3((Dn * Dn) / 256, 1, 1), 256, 0, stream>>>(Wq_w, Wqt, Dn, Dn);
  transpose_cvt<<<dim3((Dn * 2 * DKn) / 256, 1, Gn), 256, 0, stream>>>(Wkv_w, Wkvt, Dn, 2 * DKn);
  transpose_cvt<<<dim3((Dn * Dn) / 256, 1, 1), 256, 0, stream>>>(Wo_w, Wot, Dn, Dn);

  // 2) q projection: (8192x1024)@(1024x1024) -> qh heads, scale folded
  gemm_wmma<0><<<dim3(Mrows / 128, Dn / 128), 256, 0, stream>>>(
      Qbf, Wqt, Wq_b, nullptr, qhp, nullptr);

  // 3) grouped kv projection: per g, (8192x1024)@(1024x256) -> kh + vT
  gemm_wmma<1><<<dim3(Mrows / 128, (2 * DKn) / 128, Gn), 256, 0, stream>>>(
      Kbf, Wkvt, Wkv_b, nullptr, khp, vTp);

  // 4) flash attention over 32 (b,g) heads, 128 q-rows per block
  attn_flash<<<dim3(Sn / 128, Bn * Gn), 256, 0, stream>>>(qhp, khp, vTp, comb);

  // 5) output projection -> f32 result
  gemm_wmma<2><<<dim3(Mrows / 128, Dn / 128), 256, 0, stream>>>(
      comb, Wot, Wo_b, out, nullptr, nullptr);
}
